// Model6Main_72267119722887
// MI455X (gfx1250) — matrix operand order repaired and checked by reading, not compiled, not run on hardware
//
#include <hip/hip_runtime.h>

#define F_DIM 64
#define EPSV 1e-6f

typedef __attribute__((ext_vector_type(16))) _Float16 v16h;
typedef __attribute__((ext_vector_type(8)))  float    v8f;

union FragU { v16h h; unsigned int u[8]; };

__constant__ int UOFF[7] = {0, 11577, 11588, 11599, 11610, 11621, 11642};
__constant__ int ROFF[5] = {0, 4733, 4754, 4761, 4772};

__device__ __forceinline__ float wred(float v) {
    v += __shfl_xor(v, 16, 32);
    v += __shfl_xor(v, 8, 32);
    v += __shfl_xor(v, 4, 32);
    v += __shfl_xor(v, 2, 32);
    v += __shfl_xor(v, 1, 32);
    return v;
}

// Sum nf embedding rows (field-offset table offs) at feature dims [d0, d0+1].
__device__ __forceinline__ float2 gather_sum(const float* __restrict__ tab,
                                             const int* __restrict__ idx,
                                             const int* offs, int nf, int d0) {
    float2 a; a.x = 0.f; a.y = 0.f;
    for (int t = 0; t < nf; ++t) {
        const float* p = tab + (size_t)(idx[t] + offs[t]) * F_DIM + d0;
        a.x += p[0];
        a.y += p[1];
    }
    return a;
}

// softmax(cos)-weighted relu combine over 3 neighbors (lane holds 2 dims).
__device__ __forceinline__ float2 gat_combine(const float2 e[3], const float cs[3],
                                              const float* __restrict__ wgt) {
    float m = fmaxf(cs[0], fmaxf(cs[1], cs[2]));
    float s0 = __expf(cs[0] - m), s1 = __expf(cs[1] - m), s2 = __expf(cs[2] - m);
    float inv = 1.f / (s0 + s1 + s2);
    float w0 = wgt[0], w1 = wgt[1], w2 = wgt[2];
    s0 *= inv; s1 *= inv; s2 *= inv;
    float2 o;
    o.x = fmaxf(e[0].x * w0, 0.f) * s0 + fmaxf(e[1].x * w1, 0.f) * s1 + fmaxf(e[2].x * w2, 0.f) * s2;
    o.y = fmaxf(e[0].y * w0, 0.f) * s0 + fmaxf(e[1].y * w1, 0.f) * s1 + fmaxf(e[2].y * w2, 0.f) * s2;
    return o;
}

__device__ __forceinline__ float cos_from(const float2 e, const float2 c, float nb) {
    float dp = wred(e.x * c.x + e.y * c.y);
    float nn = wred(e.x * e.x + e.y * e.y);
    return dp / (fmaxf(sqrtf(nn), EPSV) * nb);
}

// ---------------------------------------------------------------------------
// Fully fused model: 16 samples per 512-thread (16-wave) workgroup.
//   A: hop-2 gathers + gat2  -> Ash[96][128] (f16 cat rows)
//   B: WMMA  Ash @ w_up2     -> C1[96][64]  (f16, stays in LDS)
//   C: hop-1 gat1 from C1    -> Ash[0..15]  (f16 cat rows); Bsh <- w_up1
//   D: WMMA  Ash @ w_up1     -> Uemb[16][64] (f32, stays in LDS)
//   E: history mean-pools + w_out projection + softmax -> out
// ---------------------------------------------------------------------------
__global__ __launch_bounds__(512, 1) void k_fused(
    const int* __restrict__ user_f,         // (N,7)
    const int* __restrict__ url_f,          // (N,5)
    const int* __restrict__ u1user_f,       // (N,3,7)
    const float* __restrict__ u1user_w,     // (N,3)
    const int* __restrict__ u1url_f,        // (N,3,5)
    const float* __restrict__ u1url_w,      // (N,3)
    const int* __restrict__ u2user_f,       // (N,3,3,7)
    const float* __restrict__ u2user_w,     // (N,3,3)
    const int* __restrict__ u2url_f,        // (N,3,3,5)
    const float* __restrict__ u2url_w,
    const int* __restrict__ r2user_f,       // (N,3,3,7)
    const float* __restrict__ r2user_w,
    const int* __restrict__ r2url_f,        // (N,3,3,5)
    const float* __restrict__ r2url_w,
    const int* __restrict__ consume_f,      // (N,20,5)
    const int* __restrict__ nurl_f,         // (N,20,5)
    const int* __restrict__ friend_f,       // (N,20,7)
    const int* __restrict__ nuser_f,        // (N,20,7)
    const float* __restrict__ user_table,
    const float* __restrict__ url_table,
    const float* __restrict__ w_up2, const float* __restrict__ b_up2,
    const float* __restrict__ w_up1, const float* __restrict__ b_up1,
    const float* __restrict__ w_out, const float* __restrict__ b_out,
    float* __restrict__ out,                // (N,2)
    int N)
{
    __shared__ _Float16 Ash[96][130];   // cat rows x K=128 (pad 2 halfs)
    __shared__ _Float16 Bsh[64][130];   // weight transposed: [n][k]
    __shared__ _Float16 C1[96][72];     // hop-2 GEMM out (pad 8 halfs)
    __shared__ float    Uemb[16][68];   // hop-1 GEMM out (pad 4 floats)

    const int tid  = threadIdx.x;
    const int wv   = tid >> 5;
    const int lane = tid & 31;
    const int d0   = lane * 2;
    const int hl   = lane & 15;
    const int hi   = lane >> 4;
    const int s    = blockIdx.x * 16 + wv;

    // ---- stage w_up2 (transposed) while phase A runs
    for (int i = tid; i < 128 * 64; i += 512) {
        int k = i >> 6, n = i & 63;
        Bsh[n][k] = (_Float16)w_up2[i];
    }

    // ================= Phase A: hop-2 =================
    if (s < N) {
        // URL branch: center = ur(user_1_url_f_list)
        for (int k = 0; k < 3; ++k) {
            size_t sk = (size_t)s * 3 + k;
            float2 c = gather_sum(url_table, u1url_f + sk * 5, ROFF, 5, d0);
            float nb = fmaxf(sqrtf(wred(c.x * c.x + c.y * c.y)), EPSV);
            float2 e[3]; float cs[3];
            const int* ib = r2user_f + sk * 3 * 7;
            for (int j = 0; j < 3; ++j) {
                e[j] = gather_sum(user_table, ib + j * 7, UOFF, 7, d0);
                cs[j] = cos_from(e[j], c, nb);
            }
            float2 g1 = gat_combine(e, cs, r2user_w + sk * 3);
            ib = r2url_f + sk * 3 * 5;
            for (int j = 0; j < 3; ++j) {
                e[j] = gather_sum(url_table, ib + j * 5, ROFF, 5, d0);
                cs[j] = cos_from(e[j], c, nb);
            }
            float2 g2 = gat_combine(e, cs, r2url_w + sk * 3);
            int row = wv * 3 + k;
            Ash[row][d0]          = (_Float16)c.x;
            Ash[row][d0 + 1]      = (_Float16)c.y;
            Ash[row][64 + d0]     = (_Float16)((g1.x + g2.x) * 0.5f);
            Ash[row][64 + d0 + 1] = (_Float16)((g1.y + g2.y) * 0.5f);
        }
        // USER branch: center = ue(user_1_user_f_list)
        for (int k = 0; k < 3; ++k) {
            size_t sk = (size_t)s * 3 + k;
            float2 c = gather_sum(user_table, u1user_f + sk * 7, UOFF, 7, d0);
            float nb = fmaxf(sqrtf(wred(c.x * c.x + c.y * c.y)), EPSV);
            float2 e[3]; float cs[3];
            const int* ib = u2user_f + sk * 3 * 7;
            for (int j = 0; j < 3; ++j) {
                e[j] = gather_sum(user_table, ib + j * 7, UOFF, 7, d0);
                cs[j] = cos_from(e[j], c, nb);
            }
            float2 g1 = gat_combine(e, cs, u2user_w + sk * 3);
            ib = u2url_f + sk * 3 * 5;
            for (int j = 0; j < 3; ++j) {
                e[j] = gather_sum(url_table, ib + j * 5, ROFF, 5, d0);
                cs[j] = cos_from(e[j], c, nb);
            }
            float2 g2 = gat_combine(e, cs, u2url_w + sk * 3);
            int row = 48 + wv * 3 + k;
            Ash[row][d0]          = (_Float16)c.x;
            Ash[row][d0 + 1]      = (_Float16)c.y;
            Ash[row][64 + d0]     = (_Float16)((g1.x + g2.x) * 0.5f);
            Ash[row][64 + d0 + 1] = (_Float16)((g1.y + g2.y) * 0.5f);
        }
    }
    __syncthreads();

    // ================= Phase B: 96x128 @ 128x64 (WMMA), result -> C1 ========
    for (int t = wv; t < 24; t += 16) {
        int mt = t >> 2, nt = t & 3;
        v8f acc = {};
        for (int kk = 0; kk < 4; ++kk) {
            FragU a, b;
#pragma unroll
            for (int j = 0; j < 8; ++j) {
                int ka = kk * 32 + (j & 3) * 2 + (j >> 2) * 16 + hi * 8; // A layout (ISA 7.12.2)
                int kb = kk * 32 + (j & 3) * 2 + (j >> 2) * 16 + hi * 8; // B layout
                a.u[j] = *(const unsigned int*)&Ash[mt * 16 + hl][ka];
                b.u[j] = *(const unsigned int*)&Bsh[nt * 16 + hl][kb];
            }
            acc = __builtin_amdgcn_wmma_f32_16x16x32_f16(false, a.h, false, b.h,
                                                         (short)0, acc, false, false);
        }
        int col = nt * 16 + hl;
        float bias = b_up2[col];
        int rbase = mt * 16 + hi * 8;
#pragma unroll
        for (int r = 0; r < 8; ++r) {
            C1[rbase + r][col] = (_Float16)fmaxf(acc[r] + bias, 0.f);
        }
    }
    __syncthreads();

    // ---- stage w_up1 (transposed); Bsh free after phase B
    for (int i = tid; i < 128 * 64; i += 512) {
        int k = i >> 6, n = i & 63;
        Bsh[n][k] = (_Float16)w_up1[i];
    }

    // ================= Phase C: hop-1 gat1 from C1 ==========================
    if (s < N) {
        float2 c = gather_sum(user_table, user_f + (size_t)s * 7, UOFF, 7, d0);
        float nb = fmaxf(sqrtf(wred(c.x * c.x + c.y * c.y)), EPSV);
        float2 e[3]; float cs[3];
        for (int j = 0; j < 3; ++j) {                 // u1url rows 0..47
            const _Float16* p = &C1[wv * 3 + j][d0];
            e[j].x = (float)p[0]; e[j].y = (float)p[1];
            cs[j] = cos_from(e[j], c, nb);
        }
        float2 gu = gat_combine(e, cs, u1url_w + (size_t)s * 3);
        for (int j = 0; j < 3; ++j) {                 // u1user rows 48..95
            const _Float16* p = &C1[48 + wv * 3 + j][d0];
            e[j].x = (float)p[0]; e[j].y = (float)p[1];
            cs[j] = cos_from(e[j], c, nb);
        }
        float2 gs = gat_combine(e, cs, u1user_w + (size_t)s * 3);
        Ash[wv][d0]          = (_Float16)c.x;
        Ash[wv][d0 + 1]      = (_Float16)c.y;
        Ash[wv][64 + d0]     = (_Float16)((gu.x + gs.x) * 0.5f);
        Ash[wv][64 + d0 + 1] = (_Float16)((gu.y + gs.y) * 0.5f);
    }
    __syncthreads();

    // ================= Phase D: 16x128 @ 128x64 (WMMA) -> Uemb ==============
    if (wv < 4) {
        int nt = wv;
        v8f acc = {};
        for (int kk = 0; kk < 4; ++kk) {
            FragU a, b;
#pragma unroll
            for (int j = 0; j < 8; ++j) {
                int ka = kk * 32 + (j & 3) * 2 + (j >> 2) * 16 + hi * 8;
                int kb = kk * 32 + (j & 3) * 2 + (j >> 2) * 16 + hi * 8;
                a.u[j] = *(const unsigned int*)&Ash[hl][ka];
                b.u[j] = *(const unsigned int*)&Bsh[nt * 16 + hl][kb];
            }
            acc = __builtin_amdgcn_wmma_f32_16x16x32_f16(false, a.h, false, b.h,
                                                         (short)0, acc, false, false);
        }
        int col = nt * 16 + hl;
        float bias = b_up1[col];
#pragma unroll
        for (int r = 0; r < 8; ++r) {
            Uemb[r + hi * 8][col] = fmaxf(acc[r] + bias, 0.f);
        }
    }
    __syncthreads();

    // ================= Phase E: history pools + final projection ============
    if (s < N) {
        float2 x[6];
#pragma unroll
        for (int b = 0; b < 4; ++b) {
            const int* lst = (b == 0) ? consume_f : (b == 1) ? nurl_f
                           : (b == 2) ? friend_f : nuser_f;
            const float* tab = (b < 2) ? url_table : user_table;
            const int* offs = (b < 2) ? ROFF : UOFF;
            int nf = (b < 2) ? 5 : 7;
            float2 a; a.x = 0.f; a.y = 0.f;
            for (int h = 0; h < 20; ++h) {
                float2 v = gather_sum(tab, lst + ((size_t)s * 20 + h) * nf, offs, nf, d0);
                a.x += v.x; a.y += v.y;
            }
            x[b].x = a.x * 0.05f; x[b].y = a.y * 0.05f;
        }
        x[4].x = Uemb[wv][d0];
        x[4].y = Uemb[wv][d0 + 1];
        x[5] = gather_sum(url_table, url_f + (size_t)s * 5, ROFF, 5, d0);

        float p0 = 0.f, p1 = 0.f;
#pragma unroll
        for (int b = 0; b < 6; ++b) {
            int i0 = b * F_DIM + d0;
            p0 += x[b].x * w_out[i0 * 2]     + x[b].y * w_out[(i0 + 1) * 2];
            p1 += x[b].x * w_out[i0 * 2 + 1] + x[b].y * w_out[(i0 + 1) * 2 + 1];
        }
        p0 = wred(p0) + b_out[0];
        p1 = wred(p1) + b_out[1];
        float m = fmaxf(p0, p1);
        float e0 = __expf(p0 - m), e1 = __expf(p1 - m);
        float inv = 1.f / (e0 + e1);
        if (lane == 0) {
            out[(size_t)s * 2]     = e0 * inv;
            out[(size_t)s * 2 + 1] = e1 * inv;
        }
    }
}

extern "C" void kernel_launch(void* const* d_in, const int* in_sizes, int n_in,
                              void* d_out, int out_size, void* d_ws, size_t ws_size,
                              hipStream_t stream) {
    const int N = in_sizes[0];                       // url_idxs (unused by model)
    const int*   user_f     = (const int*)d_in[1];
    const int*   url_f      = (const int*)d_in[2];
    const int*   u1user_f   = (const int*)d_in[3];
    const float* u1user_w   = (const float*)d_in[4];
    const int*   u1url_f    = (const int*)d_in[5];
    const float* u1url_w    = (const float*)d_in[6];
    const int*   u2user_f   = (const int*)d_in[7];
    const float* u2user_w   = (const float*)d_in[8];
    const int*   u2url_f    = (const int*)d_in[9];
    const float* u2url_w    = (const float*)d_in[10];
    const int*   r2user_f   = (const int*)d_in[11];
    const float* r2user_w   = (const float*)d_in[12];
    const int*   r2url_f    = (const int*)d_in[13];
    const float* r2url_w    = (const float*)d_in[14];
    const int*   consume_f  = (const int*)d_in[15];
    const int*   nurl_f     = (const int*)d_in[16];
    const int*   friend_f   = (const int*)d_in[17];
    const int*   nuser_f    = (const int*)d_in[18];
    const float* user_table = (const float*)d_in[19];
    const float* url_table  = (const float*)d_in[20];
    const float* w_up2      = (const float*)d_in[21];
    const float* b_up2      = (const float*)d_in[22];
    const float* w_up1      = (const float*)d_in[23];
    const float* b_up1      = (const float*)d_in[24];
    const float* w_out      = (const float*)d_in[25];
    const float* b_out      = (const float*)d_in[26];

    int nb = (N + 15) / 16;
    k_fused<<<nb, 512, 0, stream>>>(user_f, url_f, u1user_f, u1user_w, u1url_f, u1url_w,
                                    u2user_f, u2user_w, u2url_f, u2url_w,
                                    r2user_f, r2user_w, r2url_f, r2url_w,
                                    consume_f, nurl_f, friend_f, nuser_f,
                                    user_table, url_table,
                                    w_up2, b_up2, w_up1, b_up1, w_out, b_out,
                                    (float*)d_out, N);
}
